// WindowAttention_37014028157042
// MI455X (gfx1250) — matrix operand order repaired and checked by reading, not compiled, not run on hardware
//
#include <hip/hip_runtime.h>
#include <hip/hip_bf16.h>

typedef __attribute__((ext_vector_type(16))) _Float16 v16h;
typedef __attribute__((ext_vector_type(8)))  _Float16 v8h;
typedef __attribute__((ext_vector_type(8)))  float    v8f;

#define HEADS 16
#define DMODEL 1024
#define HD 64
#define WIN 256

union FragU { v16h v; v8h h8[2]; };

// A-fragment (16x32, f16): lane holds row M = lane&15.
// K pattern per lane-half: kb = kOff + (hi?8:0); elems 0..7 -> K=kb..kb+7, elems 8..15 -> K=kb+16..kb+23
__device__ __forceinline__ v16h load_fragA(const _Float16* t, int rowBase, int ld, int kOff) {
    const int lane = threadIdx.x & 31;
    const int hi = lane >> 4;
    const _Float16* p = t + (size_t)(rowBase + (lane & 15)) * ld + kOff + hi * 8;
    FragU u;
    u.h8[0] = *(const v8h*)(p);
    u.h8[1] = *(const v8h*)(p + 16);
    return u.v;
}

// B-fragment (32x16, f16) from tile stored [n][k] row-major (i.e. B[k][n] = tile[n][k]):
// lane holds col N = lane&15; kb = kOff + (hi?8:0); elems 0..7 -> K=kb..kb+7, elems 8..15 -> K=kb+16..kb+23
__device__ __forceinline__ v16h load_fragB(const _Float16* t, int rowBase, int ld, int kOff) {
    const int lane = threadIdx.x & 31;
    const int hi = lane >> 4;
    const _Float16* p = t + (size_t)(rowBase + (lane & 15)) * ld + kOff + hi * 8;
    FragU u;
    u.h8[0] = *(const v8h*)(p);
    u.h8[1] = *(const v8h*)(p + 16);
    return u.v;
}

// ---------------- LayerNorm -> f16 ----------------
__global__ __launch_bounds__(256) void ln_kernel(const float* __restrict__ x,
                                                 const float* __restrict__ g,
                                                 const float* __restrict__ b,
                                                 _Float16* __restrict__ out) {
    const int row = blockIdx.x;
    const float* xr = x + (size_t)row * DMODEL;
    float xv[4];
    float s = 0.f, s2 = 0.f;
#pragma unroll
    for (int i = 0; i < 4; ++i) {
        xv[i] = xr[threadIdx.x + i * 256];
        s += xv[i];
        s2 += xv[i] * xv[i];
    }
#pragma unroll
    for (int off = 16; off; off >>= 1) {
        s  += __shfl_xor(s,  off, 32);
        s2 += __shfl_xor(s2, off, 32);
    }
    __shared__ float rs[8], rs2[8];
    if ((threadIdx.x & 31) == 0) { rs[threadIdx.x >> 5] = s; rs2[threadIdx.x >> 5] = s2; }
    __syncthreads();
    float sum = 0.f, sum2 = 0.f;
#pragma unroll
    for (int w = 0; w < 8; ++w) { sum += rs[w]; sum2 += rs2[w]; }
    const float mean = sum * (1.0f / DMODEL);
    const float var  = sum2 * (1.0f / DMODEL) - mean * mean;
    const float inv  = rsqrtf(var + 1e-5f);
#pragma unroll
    for (int i = 0; i < 4; ++i) {
        const int c = threadIdx.x + i * 256;
        out[(size_t)row * DMODEL + c] = (_Float16)((xv[i] - mean) * inv * g[c] + b[c]);
    }
}

// ---------------- weight f32 -> f16 (layout kept [n][k]) ----------------
__global__ __launch_bounds__(256) void wconv_kernel(const float* __restrict__ w,
                                                    _Float16* __restrict__ wh, int n) {
    const int i = blockIdx.x * 256 + threadIdx.x;
    if (i < n) wh[i] = (_Float16)w[i];
}

// ---------------- rotary tables: pos 0..511 x dim 0..63 ----------------
__global__ void rope_kernel(float* __restrict__ c, float* __restrict__ s) {
    const int pos = blockIdx.x;   // 0..511
    const int d   = threadIdx.x;  // 0..63
    const float inv = __powf(10000.f, -(float)(2 * (d >> 1)) / 64.f);
    const float f = (float)pos * inv;
    c[pos * 64 + d] = __cosf(f);
    s[pos * 64 + d] = __sinf(f);
}

// ---------------- GEMM: out[m][n] = A(f16)[m][k] * Bw(f16)[n][k] + bias[n] ----------------
// workgroup tile 128(M) x 128(N), 8 waves (4x2), wave tile 32x64, K-step 64,
// double-buffered LDS with register-staged global loads (loads overlap WMMA).
template <bool OUT_F16>
__global__ __launch_bounds__(256) void gemm_kernel(const _Float16* __restrict__ A,
                                                   const _Float16* __restrict__ Bw,
                                                   const float* __restrict__ bias,
                                                   void* __restrict__ out,
                                                   int M, int N, int K) {
    extern __shared__ _Float16 sm[];  // [2][128*64 (A) + 128*64 (B)] = 64 KB
    const int m0 = blockIdx.y * 128;
    const int n0 = blockIdx.x * 128;
    const int tid = threadIdx.x;
    const int wave = tid >> 5;
    const int lane = tid & 31;
    const int wm = wave & 3;   // 4 waves along M (32 rows each)
    const int wn = wave >> 2;  // 2 waves along N (64 cols each)
    const int hi = lane >> 4, ln_ = lane & 15;

    // staging registers: per matrix 128x64 halfs = 2048 uint2 units, 8 per thread
    uint2 ra[8], rb[8];

    auto loadRegs = [&](int k0) {
#pragma unroll
        for (int i = 0; i < 8; ++i) {
            const int u = tid + i * 256;
            const int row = u >> 4, c4 = (u & 15) << 2;
            ra[i] = *(const uint2*)&A[(size_t)(m0 + row) * K + k0 + c4];
            rb[i] = *(const uint2*)&Bw[(size_t)(n0 + row) * K + k0 + c4];
        }
    };
    auto storeRegs = [&](int buf) {
        _Float16* As = sm + (size_t)buf * (2 * 128 * 64);
        _Float16* Bs = As + 128 * 64;
#pragma unroll
        for (int i = 0; i < 8; ++i) {
            const int u = tid + i * 256;
            const int row = u >> 4, c4 = (u & 15) << 2;
            *(uint2*)&As[row * 64 + c4] = ra[i];
            *(uint2*)&Bs[row * 64 + c4] = rb[i];
        }
    };

    v8f acc[2][4] = {};
    loadRegs(0);
    const int iters = K / 64;
    for (int it = 0; it < iters; ++it) {
        storeRegs(it & 1);
        __syncthreads();
        if (it + 1 < iters) loadRegs((it + 1) * 64);  // next tile in flight during WMMA
        const _Float16* As = sm + (size_t)(it & 1) * (2 * 128 * 64);
        const _Float16* Bs = As + 128 * 64;
#pragma unroll
        for (int ko = 0; ko < 64; ko += 32) {
            v16h af[2], bf[4];
            af[0] = load_fragA(As, wm * 32, 64, ko);
            af[1] = load_fragA(As, wm * 32 + 16, 64, ko);
#pragma unroll
            for (int nt = 0; nt < 4; ++nt) bf[nt] = load_fragB(Bs, wn * 64 + nt * 16, 64, ko);
#pragma unroll
            for (int mt = 0; mt < 2; ++mt)
#pragma unroll
                for (int nt = 0; nt < 4; ++nt)
                    acc[mt][nt] = __builtin_amdgcn_wmma_f32_16x16x32_f16(
                        false, af[mt], false, bf[nt], (short)0, acc[mt][nt], false, false);
        }
    }
#pragma unroll
    for (int mt = 0; mt < 2; ++mt)
#pragma unroll
        for (int nt = 0; nt < 4; ++nt) {
            const int n = n0 + wn * 64 + nt * 16 + ln_;
            const float bv = bias[n];
#pragma unroll
            for (int r = 0; r < 8; ++r) {
                const int m = m0 + wm * 32 + mt * 16 + r + 8 * hi;
                const float val = acc[mt][nt][r] + bv;
                if (OUT_F16) ((_Float16*)out)[(size_t)m * N + n] = (_Float16)val;
                else         ((float*)out)[(size_t)m * N + n] = val;
            }
        }
}

// ---------------- windowed flash attention ----------------
// one block per (b, h, bucket); 8 waves, each owns 32 query rows
__global__ __launch_bounds__(256) void attn_kernel(const _Float16* __restrict__ qp,
                                                   const _Float16* __restrict__ kp,
                                                   const _Float16* __restrict__ vp,
                                                   const int* __restrict__ cell,
                                                   const float* __restrict__ ropec,
                                                   const float* __restrict__ ropes,
                                                   float* __restrict__ osc,
                                                   int B, int L) {
    extern __shared__ char smem[];
    _Float16* Qs  = (_Float16*)smem;          // 256*64
    _Float16* Ks  = Qs + 256 * 64;            // 64*64
    _Float16* Vts = Ks + 64 * 64;             // 64*64 stored [d][slot]
    _Float16* Ps  = Vts + 64 * 64;            // 8 waves * 32*64
    int* cellq = (int*)(Ps + 8 * 32 * 64);    // 256
    int* cellk = cellq + 256;                 // 64

    const int nb = L / WIN;
    const int nbk = blockIdx.x % nb;
    const int h   = (blockIdx.x / nb) % HEADS;
    const int b   = blockIdx.x / (nb * HEADS);
    const int tid = threadIdx.x, lane = tid & 31, wave = tid >> 5;
    const int hi = lane >> 4, ln_ = lane & 15;

    // stage Q with rotary (pos = local row), one row per thread
    {
        const int r = tid;
        const size_t grow = ((size_t)b * L + (size_t)nbk * WIN + r) * DMODEL + h * HD;
#pragma unroll
        for (int d2 = 0; d2 < 32; ++d2) {
            const float x0 = (float)qp[grow + 2 * d2];
            const float x1 = (float)qp[grow + 2 * d2 + 1];
            const float cc = ropec[r * 64 + 2 * d2];
            const float ss = ropes[r * 64 + 2 * d2];
            Qs[r * 64 + 2 * d2]     = (_Float16)(x0 * cc - x1 * ss);
            Qs[r * 64 + 2 * d2 + 1] = (_Float16)(x1 * cc + x0 * ss);
        }
        cellq[r] = cell[(size_t)b * L + (size_t)nbk * WIN + r];
    }

    float m_run[2][8], l_run[2][8];
    v8f o_acc[2][4];
#pragma unroll
    for (int mt = 0; mt < 2; ++mt) {
#pragma unroll
        for (int r = 0; r < 8; ++r) { m_run[mt][r] = -1e30f; l_run[mt][r] = 0.f; }
#pragma unroll
        for (int nt = 0; nt < 4; ++nt) o_acc[mt][nt] = (v8f){};
    }

    const int kb0 = (nbk == 0) ? 4 : 0;  // bucket 0: "previous bucket" keys are pad -> fully masked
    for (int kb = kb0; kb < 8; ++kb) {
        __syncthreads();  // protect Q/Ks/Vts staging (WAR across iterations)
        {
            // stage K block (64 slots) with rotary (pos = slot index in 0..511), and V transposed
            const int s_ = tid >> 2;          // slot within block: 0..63
            const int pg = (tid & 3) * 8;     // pair-group start
            const int slot = kb * 64 + s_;
            const int j = (nbk - 1) * WIN + slot;  // absolute key position (>=0 given kb0)
            const size_t grow = ((size_t)b * L + j) * DMODEL + h * HD;
#pragma unroll
            for (int p = 0; p < 8; ++p) {
                const int d2 = pg + p;
                const float x0 = (float)kp[grow + 2 * d2];
                const float x1 = (float)kp[grow + 2 * d2 + 1];
                const float cc = ropec[slot * 64 + 2 * d2];
                const float ss = ropes[slot * 64 + 2 * d2];
                Ks[s_ * 64 + 2 * d2]     = (_Float16)(x0 * cc - x1 * ss);
                Ks[s_ * 64 + 2 * d2 + 1] = (_Float16)(x1 * cc + x0 * ss);
            }
#pragma unroll
            for (int p = 0; p < 16; ++p) {
                const int d = (tid & 3) * 16 + p;
                Vts[d * 64 + s_] = vp[grow + d];
            }
            if (tid < 64) cellk[tid] = cell[(size_t)b * L + (nbk - 1) * WIN + kb * 64 + tid];
        }
        __syncthreads();

        // S = Q * K^T (per-wave 32x64 block)
        v8f sa[2][4];
#pragma unroll
        for (int mt = 0; mt < 2; ++mt)
#pragma unroll
            for (int nt = 0; nt < 4; ++nt) sa[mt][nt] = (v8f){};
#pragma unroll
        for (int ko = 0; ko < 64; ko += 32) {
            v16h af[2], bf[4];
            af[0] = load_fragA(Qs, wave * 32, 64, ko);
            af[1] = load_fragA(Qs, wave * 32 + 16, 64, ko);
#pragma unroll
            for (int nt = 0; nt < 4; ++nt) bf[nt] = load_fragB(Ks, nt * 16, 64, ko);
#pragma unroll
            for (int mt = 0; mt < 2; ++mt)
#pragma unroll
                for (int nt = 0; nt < 4; ++nt)
                    sa[mt][nt] = __builtin_amdgcn_wmma_f32_16x16x32_f16(
                        false, af[mt], false, bf[nt], (short)0, sa[mt][nt], false, false);
        }

        // mask + online softmax; write P (f16) into this wave's LDS scratch
        const float scale = 0.125f;  // 1/sqrt(64)
        const _Float16* __restrict__ pwave_c = Ps + wave * 2048;
        _Float16* pwave = Ps + wave * 2048;
#pragma unroll
        for (int mt = 0; mt < 2; ++mt) {
#pragma unroll
            for (int r = 0; r < 8; ++r) {
                const int qrow = wave * 32 + mt * 16 + r + 8 * hi;
                const int iabs = qrow + WIN;  // positions relative to (nbk-1)*WIN
                float sv[4];
                float mb = -__builtin_inff();
#pragma unroll
                for (int nt = 0; nt < 4; ++nt) {
                    const int kslot = kb * 64 + nt * 16 + ln_;
                    float s = sa[mt][nt][r] * scale;
                    const bool valid = (iabs >= kslot) && (cellq[qrow] == cellk[nt * 16 + ln_]);
                    s = valid ? s : -__builtin_inff();
                    sv[nt] = s;
                    mb = fmaxf(mb, s);
                }
#pragma unroll
                for (int off = 1; off < 16; off <<= 1) mb = fmaxf(mb, __shfl_xor(mb, off, 16));
                const float mnew = fmaxf(m_run[mt][r], mb);
                const float sc = __expf(m_run[mt][r] - mnew);
                float rsum = 0.f;
#pragma unroll
                for (int nt = 0; nt < 4; ++nt) {
                    const float p = __expf(sv[nt] - mnew);
                    rsum += p;
                    pwave[(mt * 16 + r + 8 * hi) * 64 + nt * 16 + ln_] = (_Float16)p;
                }
#pragma unroll
                for (int off = 1; off < 16; off <<= 1) rsum += __shfl_xor(rsum, off, 16);
                l_run[mt][r] = l_run[mt][r] * sc + rsum;
                m_run[mt][r] = mnew;
#pragma unroll
                for (int nt = 0; nt < 4; ++nt) o_acc[mt][nt] *= sc;
            }
        }

        // O += P * V   (contraction over the 64 key slots; Vts stored [d][slot])
#pragma unroll
        for (int ko = 0; ko < 64; ko += 32) {
            v16h af[2], bf[4];
            af[0] = load_fragA(pwave_c, 0, 64, ko);
            af[1] = load_fragA(pwave_c, 16, 64, ko);
#pragma unroll
            for (int nt = 0; nt < 4; ++nt) bf[nt] = load_fragB(Vts, nt * 16, 64, ko);
#pragma unroll
            for (int mt = 0; mt < 2; ++mt)
#pragma unroll
                for (int nt = 0; nt < 4; ++nt)
                    o_acc[mt][nt] = __builtin_amdgcn_wmma_f32_16x16x32_f16(
                        false, af[mt], false, bf[nt], (short)0, o_acc[mt][nt], false, false);
        }
    }

    // normalize and store (merged-head layout (B, L, D) f32)
#pragma unroll
    for (int mt = 0; mt < 2; ++mt)
#pragma unroll
        for (int nt = 0; nt < 4; ++nt)
#pragma unroll
            for (int r = 0; r < 8; ++r) {
                const int qrow = wave * 32 + mt * 16 + r + 8 * hi;
                const size_t row = (size_t)b * L + (size_t)nbk * WIN + qrow;
                const int col = h * HD + nt * 16 + ln_;
                osc[row * DMODEL + col] = o_acc[mt][nt][r] / l_run[mt][r];
            }
}

extern "C" void kernel_launch(void* const* d_in, const int* in_sizes, int n_in,
                              void* d_out, int out_size, void* d_ws, size_t ws_size,
                              hipStream_t stream) {
    (void)in_sizes; (void)n_in; (void)out_size; (void)ws_size;
    const int B = 2, L = 4096, D = DMODEL;
    const int M = B * L;

    const float* q    = (const float*)d_in[0];
    const float* k    = (const float*)d_in[1];
    const float* v    = (const float*)d_in[2];
    const int*   cell = (const int*)d_in[3];
    const float* lnw[4] = {(const float*)d_in[4],  (const float*)d_in[8],
                           (const float*)d_in[12], (const float*)d_in[16]};
    const float* lnb[4] = {(const float*)d_in[5],  (const float*)d_in[9],
                           (const float*)d_in[13], (const float*)d_in[17]};
    const float* W[4]   = {(const float*)d_in[6],  (const float*)d_in[10],
                           (const float*)d_in[14], (const float*)d_in[18]};
    const float* bs[4]  = {(const float*)d_in[7],  (const float*)d_in[11],
                           (const float*)d_in[15], (const float*)d_in[19]};
    const float* xin[3] = {q, k, v};

    char* ws = (char*)d_ws;
    _Float16* xn  = (_Float16*)(ws);                       // 16 MB (reused)
    _Float16* wh  = (_Float16*)(ws + (16ull << 20));       //  2 MB (reused)
    _Float16* prj[3] = {(_Float16*)(ws + (18ull << 20)),   // q proj 16 MB
                        (_Float16*)(ws + (34ull << 20)),   // k proj 16 MB
                        (_Float16*)(ws + (50ull << 20))};  // v proj 16 MB
    float* osc = (float*)(ws + (66ull << 20));             // 32 MB attention out
    float* rc  = (float*)(ws + (98ull << 20));             // 128 KB
    float* rs  = (float*)(ws + (98ull << 20) + 512 * 64 * sizeof(float));

    rope_kernel<<<512, 64, 0, stream>>>(rc, rs);

    const size_t gemmShmem = 2ull * 2 * 128 * 64 * sizeof(_Float16);  // 64 KB
    dim3 gGemm(D / 128, M / 128);
    for (int t = 0; t < 3; ++t) {
        ln_kernel<<<M, 256, 0, stream>>>(xin[t], lnw[t], lnb[t], xn);
        wconv_kernel<<<(D * D) / 256, 256, 0, stream>>>(W[t], wh, D * D);
        gemm_kernel<true><<<gGemm, 256, gemmShmem, stream>>>(xn, wh, bs[t], prj[t], M, D, D);
    }

    const size_t shmem = (256 * 64 + 64 * 64 + 64 * 64 + 8 * 32 * 64) * sizeof(_Float16)
                       + (256 + 64) * sizeof(int);  // 83200 B
    attn_kernel<<<B * HEADS * (L / WIN), 256, shmem, stream>>>(
        prj[0], prj[1], prj[2], cell, rc, rs, osc, B, L);

    ln_kernel<<<M, 256, 0, stream>>>(osc, lnw[3], lnb[3], xn);
    wconv_kernel<<<(D * D) / 256, 256, 0, stream>>>(W[3], wh, D * D);
    gemm_kernel<false><<<gGemm, 256, gemmShmem, stream>>>(xn, wh, bs[3], (float*)d_out, M, D, D);
}
